// TinyMamba3Multi_44848048505423
// MI455X (gfx1250) — matrix operand order repaired and checked by reading, not compiled, not run on hardware
//
#include <hip/hip_runtime.h>
#include <hip/hip_bf16.h>
#include <stdint.h>

// TinyMamba3Multi on gfx1250: chunked (SSD) selective-scan.
// - all GEMMs via v_wmma_f32_16x16x32_f16 (f16 operands, f32 accum)
// - f16 activation tensors in HBM (halved traffic, WMMA-native)
// - async global->LDS staging (global_load_async_to_lds_b128 + s_wait_asynccnt)
// - fast silu via v_rcp_f32 instead of IEEE divide

#define B_  16
#define L_  8192
#define IN_ 57
#define DM  64
#define DS  32
#define HD  32
#define NL  2
#define OUT 6
#define DI  128
#define NH  4
#define DPROJ 324
#define T_  (B_ * L_)
#define CH  64
#define NCH (L_ / CH)

typedef __attribute__((ext_vector_type(16))) _Float16 v16h;
typedef __attribute__((ext_vector_type(8)))  float    v8f;
typedef _Float16 f16;

__device__ __forceinline__ float siluf(float x) {
  // x * sigmoid(x) with fast v_rcp_f32 (no IEEE divide expansion)
  return x * __builtin_amdgcn_rcpf(1.0f + __expf(-x));
}
__device__ __forceinline__ float softplusf(float x) {
  return (x > 20.0f) ? x : __logf(1.0f + __expf(x));
}

// ---- WMMA fragment helpers (layouts per CDNA5 ISA 7.12.2) -----------------
// A (16x32 f16, MxK), row-major source with leading dim `ld` (halfs).
__device__ __forceinline__ v16h frag_a(const f16* src, int ld, int lane) {
  int m = lane & 15, hl = lane >> 4;
  const uint32_t* row = reinterpret_cast<const uint32_t*>(src + m * ld + hl * 8);
  union { v16h h; uint32_t u[8]; } r;
#pragma unroll
  for (int i = 0; i < 8; ++i) r.u[i] = row[(i < 4) ? i : (i + 4)];
  return r.h;
}
// B (32x16 f16, KxN), K-contiguous (column-major by N) source, leading dim `ld`.
__device__ __forceinline__ v16h frag_b(const f16* src, int ld, int lane) {
  int c = lane & 15, kh = lane >> 4;
  const uint32_t* col = reinterpret_cast<const uint32_t*>(src + c * ld + kh * 8);
  union { v16h h; uint32_t u[8]; } r;
#pragma unroll
  for (int i = 0; i < 8; ++i) r.u[i] = col[(i < 4) ? i : (i + 4)];
  return r.h;
}
__device__ __forceinline__ v8f wmma_f16(v16h a, v16h b, v8f c) {
  return __builtin_amdgcn_wmma_f32_16x16x32_f16(false, a, false, b, (short)0, c, false, false);
}

// ---- async global->LDS copy of one contiguous 4KB f16 tile (one wave) -----
// ISA 10.? GLOBAL_LOAD_ASYNC_TO_LDS_B128 (GVS mode: saddr + 32-bit voffset);
// 32 lanes x 16B = 512B per issue, 8 issues = 4KB. Tracked by ASYNCcnt.
__device__ __forceinline__ uint32_t lds_addr_of(const void* p) {
  // low 32 bits of a generic LDS address == offset within workgroup LDS
  return (uint32_t)(uintptr_t)p;
}
__device__ __forceinline__ void async_copy_4k(uint32_t lds_base, const void* gbase,
                                              uint32_t goff, int lane) {
#pragma unroll
  for (int i = 0; i < 8; ++i) {
    uint32_t off = (uint32_t)(lane * 16 + i * 512);
    uint32_t lds = lds_base + off;
    uint32_t vo  = goff + off;
    asm volatile("global_load_async_to_lds_b128 %0, %1, %2"
                 :: "v"(lds), "v"(vo), "s"(gbase) : "memory");
  }
}
__device__ __forceinline__ void async_wait0() {
  asm volatile("s_wait_asynccnt 0x0" ::: "memory");
}

// ---- weight pre-conversion: fp32 row-major -> f16 K-contiguous ------------
__global__ void k_convw(const float* __restrict__ inW, const float* __restrict__ outW,
                        f16* __restrict__ wtA, f16* __restrict__ wtO) {
  int idx = blockIdx.x * blockDim.x + threadIdx.x;
  int stride = gridDim.x * blockDim.x;
  const int nA = NL * DPROJ * DM;
  for (int i = idx; i < nA; i += stride) {
    int l = i / (DPROJ * DM);
    int r = i % (DPROJ * DM);
    int n = r / DM, k = r % DM;
    wtA[(size_t)l * DPROJ * DM + n * DM + k] =
        (f16)inW[(size_t)l * DM * DPROJ + (size_t)k * DPROJ + n];
  }
  const int nO = NL * DM * DI;
  for (int i = idx; i < nO; i += stride) {
    int l = i / (DM * DI);
    int r = i % (DM * DI);
    int n = r / DI, k = r % DI;
    wtO[(size_t)l * DM * DI + n * DI + k] =
        (f16)outW[(size_t)l * DI * DM + (size_t)k * DM + n];
  }
}

// ---- h = x @ W_in + b_in (K=57: VALU, tiny fraction of total flops) -------
__global__ void k_linear_in(const float* __restrict__ x, const float* __restrict__ Win,
                            const float* __restrict__ bin, float* __restrict__ h) {
  __shared__ float xr[4][IN_];
  int tok0 = blockIdx.x * 4;
  int tid = threadIdx.x;
  for (int i = tid; i < 4 * IN_; i += 256) {
    int tt = i / IN_, k = i % IN_;
    xr[tt][k] = x[(size_t)(tok0 + tt) * IN_ + k];
  }
  __syncthreads();
  int tt = tid >> 6, d = tid & 63;
  float acc = bin[d];
  for (int k = 0; k < IN_; ++k) acc += xr[tt][k] * Win[k * DM + d];
  h[(size_t)(tok0 + tt) * DM + d] = acc;
}

// ---- zxbcdt = h @ W + b (cols 0..319), fused silu, f16 outputs, WMMA ------
#define LDH 72
__global__ void k_proj(const float* __restrict__ h, const f16* __restrict__ wtA,
                       const float* __restrict__ bias,
                       f16* __restrict__ zs, f16* __restrict__ xs,
                       f16* __restrict__ Bst, f16* __restrict__ Cst) {
  __shared__ f16 hH[128 * LDH];
  int tok0 = blockIdx.x * 128;
  int tid = threadIdx.x;
  for (int i = tid; i < 128 * DM; i += 256) {
    int r = i >> 6, k = i & 63;
    hH[r * LDH + k] = (f16)h[(size_t)(tok0 + r) * DM + k];
  }
  __syncthreads();
  int wave = tid >> 5, lane = tid & 31;
  int r0 = wave * 16;
  int nloc = lane & 15, mb = (lane >> 4) * 8;
  for (int nt = 0; nt < 20; ++nt) {
    v8f acc = {};
#pragma unroll
    for (int kt = 0; kt < 2; ++kt) {
      v16h a = frag_a(hH + r0 * LDH + kt * 32, LDH, lane);
      v16h b = frag_b(wtA + (nt * 16) * DM + kt * 32, DM, lane);
      acc = wmma_f16(a, b, acc);
    }
    int col = nt * 16 + nloc;
    float bv = bias[col];
#pragma unroll
    for (int r = 0; r < 8; ++r) {
      size_t t = (size_t)tok0 + r0 + mb + r;
      float v = acc[r] + bv;
      f16 sv = (f16)siluf(v);
      if (col < DI)                 zs[t * DI + col] = sv;
      else if (col < 2 * DI)        xs[t * DI + (col - DI)] = sv;
      else if (col < 2 * DI + DS)   Bst[t * DS + (col - 2 * DI)] = sv;
      else                          Cst[t * DS + (col - 2 * DI - DS)] = sv;
    }
  }
}

// ---- dt tail (cols 320..323): softplus(dt + bias), dA = -exp(A_log)*dt ----
__global__ void k_dt(const float* __restrict__ h, const float* __restrict__ W,
                     const float* __restrict__ bias, const float* __restrict__ dt_bias,
                     const float* __restrict__ A_log,
                     float* __restrict__ dtb, float* __restrict__ dAb) {
  __shared__ float hr[64][65];
  __shared__ float wc[64][4];
  int tok0 = blockIdx.x * 64;
  int tid = threadIdx.x;
  for (int i = tid; i < 64 * 64; i += 256) {
    int r = i >> 6, k = i & 63;
    hr[r][k] = h[(size_t)(tok0 + r) * DM + k];
  }
  { int k = tid >> 2, hh = tid & 3; wc[k][hh] = W[k * DPROJ + 320 + hh]; }
  __syncthreads();
  int r = tid >> 2, hh = tid & 3;
  float acc = bias[320 + hh] + dt_bias[hh];
  for (int k = 0; k < 64; ++k) acc += hr[r][k] * wc[k][hh];
  float dt = softplusf(acc);
  size_t t = (size_t)tok0 + r;
  dtb[t * NH + hh] = dt;
  dAb[t * NH + hh] = -__expf(A_log[hh]) * dt;
}

// ---- intra-chunk: G = C@B^T, mask, Y_intra = (G.L)@U, Sc = Ud^T@B ---------
#define LD32 40
#define LDU  72
__global__ void k_intra(const f16* __restrict__ xs, const f16* __restrict__ Bst,
                        const f16* __restrict__ Cst, const float* __restrict__ dtb,
                        const float* __restrict__ dAb,
                        float* __restrict__ Y, float* __restrict__ Sc,
                        float* __restrict__ Pc) {
  __shared__ float cumA[CH];
  __shared__ __attribute__((aligned(16))) f16 csH[CH * DS];  // [s][n] packed, async-filled
  __shared__ f16 bsH[CH * LD32];   // [s][n]
  __shared__ f16 bsT[HD * LDU];    // [n][s]
  __shared__ f16 uT [HD * LDU];    // [p][s]  u = dt*x
  __shared__ f16 udT[HD * LDU];    // [p][s]  decayed u
  __shared__ f16 gmH[CH * LDU];    // [t][s]

  int idx = blockIdx.x;
  int ch = idx % NCH;
  int hh = (idx / NCH) % NH;
  int b  = idx / (NCH * NH);
  size_t t0 = (size_t)b * L_ + (size_t)ch * CH;
  int lane = threadIdx.x;

  // async-stage C chunk tile (contiguous 4KB f16) into LDS while VALU fills run
  async_copy_4k(lds_addr_of(csH), Cst, (uint32_t)(t0 * DS * sizeof(f16)), lane);

  for (int s = lane; s < CH; s += 32) cumA[s] = dAb[(t0 + s) * NH + hh];
  __syncthreads();
  if (lane == 0) { float a = 0.f; for (int s = 0; s < CH; ++s) { a += cumA[s]; cumA[s] = a; } }
  __syncthreads();
  float cl = cumA[CH - 1];
  for (int i = lane; i < CH * 32; i += 32) {
    int s = i >> 5, p = i & 31;
    size_t t = t0 + s;
    float xv = (float)xs[t * DI + hh * HD + p];
    float dv = dtb[t * NH + hh];
    float u  = dv * xv;
    uT [p * LDU + s] = (f16)u;
    udT[p * LDU + s] = (f16)(u * __expf(cl - cumA[s]));
    f16 bvh = Bst[t * DS + p];
    bsH[s * LD32 + p] = bvh;
    bsT[p * LDU + s]  = bvh;
  }
  async_wait0();
  __syncthreads();

  int nloc = lane & 15, mb = (lane >> 4) * 8;
  // G with causal decay mask -> gmH
  for (int tt = 0; tt < 4; ++tt) {
    v16h a = frag_a(csH + (tt * 16) * DS, DS, lane);
    for (int st = 0; st < 4; ++st) {
      v16h bb = frag_b(bsH + (st * 16) * LD32, LD32, lane);
      v8f acc = {};
      acc = wmma_f16(a, bb, acc);
      int s = st * 16 + nloc;
#pragma unroll
      for (int r = 0; r < 8; ++r) {
        int t = tt * 16 + mb + r;
        float v = (s <= t) ? acc[r] * __expf(cumA[t] - cumA[s]) : 0.f;
        gmH[t * LDU + s] = (f16)v;
      }
    }
  }
  __syncthreads();
  // Y_intra = gm @ U   (store to Y, finished by k_inter)
  for (int tt = 0; tt < 4; ++tt) {
    for (int pt = 0; pt < 2; ++pt) {
      v8f acc = {};
#pragma unroll
      for (int kt = 0; kt < 2; ++kt) {
        v16h a  = frag_a(gmH + (tt * 16) * LDU + kt * 32, LDU, lane);
        v16h bb = frag_b(uT  + (pt * 16) * LDU + kt * 32, LDU, lane);
        acc = wmma_f16(a, bb, acc);
      }
      int p = pt * 16 + nloc;
#pragma unroll
      for (int r = 0; r < 8; ++r) {
        size_t t = t0 + tt * 16 + mb + r;
        Y[t * DI + hh * HD + p] = acc[r];
      }
    }
  }
  // Sc = Ud^T @ B  (chunk-local state contribution)
  size_t scbase = (((size_t)b * NH + hh) * NCH + ch) * (HD * DS);
  for (int mt = 0; mt < 2; ++mt) {
    for (int nt = 0; nt < 2; ++nt) {
      v8f acc = {};
#pragma unroll
      for (int kt = 0; kt < 2; ++kt) {
        v16h a  = frag_a(udT + (mt * 16) * LDU + kt * 32, LDU, lane);
        v16h bb = frag_b(bsT + (nt * 16) * LDU + kt * 32, LDU, lane);
        acc = wmma_f16(a, bb, acc);
      }
      int n = nt * 16 + nloc;
#pragma unroll
      for (int r = 0; r < 8; ++r) {
        int p = mt * 16 + mb + r;
        Sc[scbase + p * DS + n] = acc[r];
      }
    }
  }
  if (lane == 0) Pc[((size_t)b * NH + hh) * NCH + ch] = __expf(cl);
}

// ---- serial chunk-state recurrence (128 steps, tiny). Sc -> S_prev --------
__global__ void k_scan(float* __restrict__ Sc, const float* __restrict__ Pc) {
  int bh = blockIdx.x;       // B_*NH
  int tid = threadIdx.x;     // 256, each owns 4 of 1024 state elems
  float4 s = {0.f, 0.f, 0.f, 0.f};
  for (int ch = 0; ch < NCH; ++ch) {
    size_t base = ((size_t)bh * NCH + ch) * (HD * DS);
    float4* ptr = reinterpret_cast<float4*>(Sc + base) + tid;
    __builtin_prefetch(ptr + 256, 0, 0);        // next chunk (global_prefetch_b8)
    float4 c = *ptr;
    *ptr = s;                                   // store S_prev for this chunk
    float p = Pc[(size_t)bh * NCH + ch];
    s.x = p * s.x + c.x; s.y = p * s.y + c.y;
    s.z = p * s.z + c.z; s.w = p * s.w + c.w;
  }
}

// ---- inter-chunk Y += P_t * C@S_prev^T, + D*x, gate by silu(z) -> Yg(f16) -
__global__ void k_inter(const f16* __restrict__ xs, const f16* __restrict__ zs,
                        const f16* __restrict__ Cst, const float* __restrict__ dAb,
                        const float* __restrict__ Sprev, const float* __restrict__ Dp,
                        const float* __restrict__ Y, f16* __restrict__ Yg) {
  __shared__ float cumA[CH];
  __shared__ __attribute__((aligned(16))) f16 csH[CH * DS];  // async-filled
  __shared__ f16 sH[HD * LD32];    // [p][n]
  int idx = blockIdx.x;
  int ch = idx % NCH;
  int hh = (idx / NCH) % NH;
  int b  = idx / (NCH * NH);
  size_t t0 = (size_t)b * L_ + (size_t)ch * CH;
  int lane = threadIdx.x;
  float Dv = Dp[hh];

  async_copy_4k(lds_addr_of(csH), Cst, (uint32_t)(t0 * DS * sizeof(f16)), lane);

  for (int s = lane; s < CH; s += 32) cumA[s] = dAb[(t0 + s) * NH + hh];
  __syncthreads();
  if (lane == 0) { float a = 0.f; for (int s = 0; s < CH; ++s) { a += cumA[s]; cumA[s] = a; } }
  __syncthreads();
  size_t sbase = (((size_t)b * NH + hh) * NCH + ch) * (HD * DS);
  for (int i = lane; i < HD * DS; i += 32) {
    int p = i >> 5, n = i & 31;
    sH[p * LD32 + n] = (f16)Sprev[sbase + p * DS + n];
  }
  async_wait0();
  __syncthreads();
  int nloc = lane & 15, mb = (lane >> 4) * 8;
  for (int tt = 0; tt < 4; ++tt) {
    v16h a = frag_a(csH + (tt * 16) * DS, DS, lane);
    for (int pt = 0; pt < 2; ++pt) {
      v16h bb = frag_b(sH + (pt * 16) * LD32, LD32, lane);
      v8f acc = {};
      acc = wmma_f16(a, bb, acc);
      int p = pt * 16 + nloc;
#pragma unroll
      for (int r = 0; r < 8; ++r) {
        int tl = tt * 16 + mb + r;
        size_t di = (t0 + tl) * DI + hh * HD + p;
        float yv = Y[di] + __expf(cumA[tl]) * acc[r] + Dv * (float)xs[di];
        Yg[di] = (f16)(yv * (float)zs[di]);    // zs already silu'd
      }
    }
  }
}

// ---- h += Ygated @ Wo + bo (WMMA, K=128) ----------------------------------
#define LDY 136
__global__ void k_outproj(const f16* __restrict__ Yg, const f16* __restrict__ wtO,
                          const float* __restrict__ bo, float* __restrict__ h) {
  __shared__ f16 yH[128 * LDY];
  int tok0 = blockIdx.x * 128;
  int tid = threadIdx.x;
  for (int i = tid; i < 128 * DI; i += 256) {
    int r = i >> 7, k = i & 127;
    yH[r * LDY + k] = Yg[(size_t)(tok0 + r) * DI + k];
  }
  __syncthreads();
  int wave = tid >> 5, lane = tid & 31;
  int r0 = wave * 16;
  int nloc = lane & 15, mb = (lane >> 4) * 8;
  for (int nt = 0; nt < 4; ++nt) {
    v8f acc = {};
#pragma unroll
    for (int kt = 0; kt < 4; ++kt) {
      v16h a  = frag_a(yH + r0 * LDY + kt * 32, LDY, lane);
      v16h bb = frag_b(wtO + (nt * 16) * DI + kt * 32, DI, lane);
      acc = wmma_f16(a, bb, acc);
    }
    int col = nt * 16 + nloc;
    float bv = bo[col];
#pragma unroll
    for (int r = 0; r < 8; ++r) {
      size_t t = (size_t)tok0 + r0 + mb + r;
      h[t * DM + col] += acc[r] + bv;
    }
  }
}

// ---- mean-pool over L, classifier -----------------------------------------
__global__ void k_pool(const float* __restrict__ h, float* __restrict__ pooled) {
  __shared__ float red[256];
  int bd = blockIdx.x;
  int b = bd >> 6, d = bd & 63;
  int tid = threadIdx.x;
  float s = 0.f;
  for (int l = tid; l < L_; l += 256)
    s += h[((size_t)b * L_ + l) * DM + d];
  red[tid] = s;
  __syncthreads();
  for (int o = 128; o > 0; o >>= 1) {
    if (tid < o) red[tid] += red[tid + o];
    __syncthreads();
  }
  if (tid == 0) pooled[b * DM + d] = red[0] * (1.0f / L_);
}

__global__ void k_cls(const float* __restrict__ pooled, const float* __restrict__ W,
                      const float* __restrict__ bias, float* __restrict__ out) {
  int tid = threadIdx.x;
  if (tid >= B_ * OUT) return;
  int b = tid / OUT, o = tid % OUT;
  float acc = bias[o];
  for (int d = 0; d < DM; ++d) acc += pooled[b * DM + d] * W[d * OUT + o];
  out[b * OUT + o] = acc;
}

extern "C" void kernel_launch(void* const* d_in, const int* in_sizes, int n_in,
                              void* d_out, int out_size, void* d_ws, size_t ws_size,
                              hipStream_t stream) {
  (void)in_sizes; (void)n_in; (void)out_size; (void)ws_size;
  const float* x       = (const float*)d_in[0];
  const float* W_in    = (const float*)d_in[1];
  const float* b_in    = (const float*)d_in[2];
  const float* in_W    = (const float*)d_in[3];
  const float* in_b    = (const float*)d_in[4];
  const float* A_log   = (const float*)d_in[5];
  const float* Dp      = (const float*)d_in[6];
  const float* dt_bias = (const float*)d_in[7];
  const float* out_W   = (const float*)d_in[8];
  const float* out_b   = (const float*)d_in[9];
  const float* cls_W   = (const float*)d_in[10];
  const float* cls_b   = (const float*)d_in[11];
  float* out = (float*)d_out;

  float* base = (float*)d_ws;
  size_t off = 0;
  auto take = [&](size_t nfloats) -> float* {
    float* p = base + off;
    off += (nfloats + 63) & ~size_t(63);
    return p;
  };
  float* h    = take((size_t)T_ * DM);
  f16*   xs   = (f16*)take((size_t)T_ * DI / 2);
  f16*   zs   = (f16*)take((size_t)T_ * DI / 2);
  f16*   Bst  = (f16*)take((size_t)T_ * DS / 2);
  f16*   Cst  = (f16*)take((size_t)T_ * DS / 2);
  float* dtb  = take((size_t)T_ * NH);
  float* dAb  = take((size_t)T_ * NH);
  float* Yb   = take((size_t)T_ * DI);
  f16*   Ygb  = (f16*)take((size_t)T_ * DI / 2);
  float* Scb  = take((size_t)B_ * NH * NCH * HD * DS);
  float* Pcb  = take((size_t)B_ * NH * NCH);
  float* pool = take((size_t)B_ * DM);
  f16* wtA = (f16*)take((size_t)NL * DPROJ * DM / 2 + 64);
  f16* wtO = (f16*)take((size_t)NL * DM * DI / 2 + 64);

  k_convw<<<64, 256, 0, stream>>>(in_W, out_W, wtA, wtO);
  k_linear_in<<<T_ / 4, 256, 0, stream>>>(x, W_in, b_in, h);

  for (int l = 0; l < NL; ++l) {
    k_proj<<<T_ / 128, 256, 0, stream>>>(h, wtA + (size_t)l * DPROJ * DM,
                                         in_b + (size_t)l * DPROJ, zs, xs, Bst, Cst);
    k_dt<<<T_ / 64, 256, 0, stream>>>(h, in_W + (size_t)l * DM * DPROJ,
                                      in_b + (size_t)l * DPROJ,
                                      dt_bias + l * NH, A_log + l * NH, dtb, dAb);
    k_intra<<<B_ * NH * NCH, 32, 0, stream>>>(xs, Bst, Cst, dtb, dAb, Yb, Scb, Pcb);
    k_scan<<<B_ * NH, 256, 0, stream>>>(Scb, Pcb);
    k_inter<<<B_ * NH * NCH, 32, 0, stream>>>(xs, zs, Cst, dAb, Scb, Dp + l * NH,
                                              Yb, Ygb);
    k_outproj<<<T_ / 128, 256, 0, stream>>>(Ygb, wtO + (size_t)l * DM * DI,
                                            out_b + (size_t)l * DM, h);
  }
  k_pool<<<B_ * DM, 256, 0, stream>>>(h, pool);
  k_cls<<<1, 128, 0, stream>>>(pool, cls_W, cls_b, out);
}
